// MauronSpatialGNN_84043920048185
// MI455X (gfx1250) — matrix operand order repaired and checked by reading, not compiled, not run on hardware
//
#include <hip/hip_runtime.h>

#define N_NODES 15000
#define N_EDGES 240000
#define ETOT    (N_EDGES + N_NODES)
#define DIN     2000
#define HID     128
#define EMBD    64
#define NCLS    33
#define NLAY    3
#define NHEAD   4
#define NEGS    0.2f
#define EPSV    1e-5f

typedef __attribute__((ext_vector_type(16))) __bf16 v16bf;
typedef __attribute__((ext_vector_type(8)))  float  v8f;

__device__ __forceinline__ float gelu_exact(float x) {
    return 0.5f * x * (1.0f + erff(x * 0.7071067811865476f));
}
__device__ __forceinline__ unsigned f2ord(float f) {
    unsigned u = __float_as_uint(f);
    return (u & 0x80000000u) ? ~u : (u | 0x80000000u);
}
__device__ __forceinline__ float ord2f(unsigned k) {
    unsigned u = (k & 0x80000000u) ? (k ^ 0x80000000u) : ~k;
    return __uint_as_float(u);
}
__device__ __forceinline__ unsigned pack_bf16(float a, float b) {
    union { __bf16 h[2]; unsigned u; } p;
    p.h[0] = (__bf16)a; p.h[1] = (__bf16)b;
    return p.u;
}

// ---------------------------------------------------------------------------
// bf16 WMMA GEMM: C[M,Nc] = A[M,K] @ B[K,Nc] + bias.  fp32 in/out.
// Block = 128 threads (4 waves); block tile 64x64; wave tile 16x64.
// LDS tiles stored pre-swizzled in WMMA fragment layout (ISA 7.12.2) so each
// operand is fetched with ds_load_b128 pairs. Double-buffered LDS (1 barrier
// per K-step); block-uniform fast path for full tiles (unpredicated b128/b64
// staging loads). Requires K % 4 == 0 (call sites: 2000 / 128 / 64).
// ---------------------------------------------------------------------------
__launch_bounds__(128)
__global__ void k_gemm_wmma(const float* __restrict__ A, const float* __restrict__ B,
                            const float* __restrict__ bias, float* __restrict__ C,
                            int M, int K, int Nc)
{
    __shared__ v16bf sAf[2][4][32];   // [buf][wave][lane] A fragment (16x32/wave)
    __shared__ v16bf sBf[2][4][32];   // [buf][col-tile][lane] B fragment (32x16)
    const int tid  = threadIdx.x;
    const int wave = tid >> 5;
    const int lane = tid & 31;
    const int rowBase = blockIdx.x * 64;
    const int colBase = blockIdx.y * 64;

    // element (m,k) of A tile: lane = m + 16*((k>>3)&1), i = (k&7)|(((k>>4)&1)<<3)
    auto writeA = [&](int b, int r, int k, float4 v) {
        int wv = r >> 4, m = r & 15;
        int ln = m + 16 * ((k >> 3) & 1);
        int ii = (k & 7) | (((k >> 4) & 1) << 3);     // in {0,4,8,12} for k%4==0
        uint2 pk;
        pk.x = pack_bf16(v.x, v.y);
        pk.y = pack_bf16(v.z, v.w);
        *(uint2*)((unsigned*)&sAf[b][wv][ln] + (ii >> 1)) = pk;
    };
    // element (k,n) of B tile: lane = n + 16*((k>>3)&1), i = (k&7)|(((k>>4)&1)<<3)
    auto writeB = [&](int b, int cn, int k2, float b00, float b01, float b10, float b11) {
        int ct = cn >> 4, n = cn & 15;
        int hh = (k2 >> 3) & 1;
        int slot = ((k2 & 7) | (((k2 >> 4) & 1) << 3)) >> 1;
        ((unsigned*)&sBf[b][ct][n + 16 * hh])[slot]     = pack_bf16(b00, b10);
        ((unsigned*)&sBf[b][ct][n + 1 + 16 * hh])[slot] = pack_bf16(b01, b11);
    };

    auto stageA = [&](int k0s, int b) {
        if (rowBase + 64 <= M && k0s + 32 <= K) {         // uniform fast path
            #pragma unroll
            for (int j = 0; j < 4; ++j) {
                int q = tid + j * 128;                    // 0..511 quads
                int r = q >> 3, k = (q & 7) * 4;
                float4 v = *(const float4*)(A + (size_t)(rowBase + r) * K + (k0s + k));
                writeA(b, r, k, v);
            }
        } else {
            #pragma unroll
            for (int j = 0; j < 4; ++j) {
                int q = tid + j * 128;
                int r = q >> 3, k = (q & 7) * 4;
                int gr = rowBase + r, gk = k0s + k;
                float4 v = {0.f, 0.f, 0.f, 0.f};
                if (gr < M && gk < K)                     // K%4==0 => quad safe
                    v = *(const float4*)(A + (size_t)gr * K + gk);
                writeA(b, r, k, v);
            }
        }
    };
    auto stageB = [&](int k0s, int b) {
        if (colBase + 64 <= Nc && k0s + 32 <= K) {        // uniform fast path (Nc even here)
            #pragma unroll
            for (int j = 0; j < 4; ++j) {
                int q = tid + j * 128;                    // 0..511 2x2 tiles
                int cn = (q >> 4) * 2, k2 = (q & 15) * 2;
                const float* p0 = B + (size_t)(k0s + k2) * Nc + colBase + cn;
                float2 r0 = *(const float2*)p0;
                float2 r1 = *(const float2*)(p0 + Nc);
                writeB(b, cn, k2, r0.x, r0.y, r1.x, r1.y);
            }
        } else {
            #pragma unroll
            for (int j = 0; j < 4; ++j) {
                int q = tid + j * 128;
                int cn = (q >> 4) * 2, k2 = (q & 15) * 2;
                int gk = k0s + k2, gc = colBase + cn;
                float b00 = 0.f, b01 = 0.f, b10 = 0.f, b11 = 0.f;
                if (gk < K) {                             // K even => gk+1 < K
                    if (gc < Nc)     { b00 = B[(size_t)gk * Nc + gc];     b10 = B[(size_t)(gk + 1) * Nc + gc]; }
                    if (gc + 1 < Nc) { b01 = B[(size_t)gk * Nc + gc + 1]; b11 = B[(size_t)(gk + 1) * Nc + gc + 1]; }
                }
                writeB(b, cn, k2, b00, b01, b10, b11);
            }
        }
    };

    v8f acc[4] = {};
    stageA(0, 0);
    stageB(0, 0);
    __syncthreads();

    int p = 0;
    for (int k0 = 0; k0 < K; k0 += 32) {
        int kn = k0 + 32;
        if (kn < K) {                   // stage next K-step into alternate buffer
            stageA(kn, p ^ 1);
            stageB(kn, p ^ 1);
        }
        v16bf af  = sAf[p][wave][lane];
        v16bf bf0 = sBf[p][0][lane];
        v16bf bf1 = sBf[p][1][lane];
        v16bf bf2 = sBf[p][2][lane];
        v16bf bf3 = sBf[p][3][lane];
        acc[0] = __builtin_amdgcn_wmma_f32_16x16x32_bf16(false, af, false, bf0, (short)0, acc[0], false, false);
        acc[1] = __builtin_amdgcn_wmma_f32_16x16x32_bf16(false, af, false, bf1, (short)0, acc[1], false, false);
        acc[2] = __builtin_amdgcn_wmma_f32_16x16x32_bf16(false, af, false, bf2, (short)0, acc[2], false, false);
        acc[3] = __builtin_amdgcn_wmma_f32_16x16x32_bf16(false, af, false, bf3, (short)0, acc[3], false, false);
        __syncthreads();
        p ^= 1;
    }

    // D layout: VGPR r -> M = r + 8*(lane>>4), N = lane&15
    const int hh = lane >> 4;
    const int nn = lane & 15;
    #pragma unroll
    for (int ct = 0; ct < 4; ++ct) {
        int gc = colBase + ct * 16 + nn;
        if (gc < Nc) {
            float bv = bias[gc];
            #pragma unroll
            for (int r = 0; r < 8; ++r) {
                int gm = rowBase + wave * 16 + 8 * hh + r;
                if (gm < M) C[(size_t)gm * Nc + gc] = acc[ct][r] + bv;
            }
        }
    }
}

// ---------------------------------------------------------------------------
// LayerNorm (+ optional exact GELU), one wave per row.
// ---------------------------------------------------------------------------
template<int WIDTH, bool DOGELU>
__launch_bounds__(128)
__global__ void k_ln(const float* __restrict__ in, const float* __restrict__ g,
                     const float* __restrict__ b, float* __restrict__ out, int n)
{
    const int wave = threadIdx.x >> 5;
    const int lane = threadIdx.x & 31;
    const int row  = blockIdx.x * 4 + wave;
    if (row >= n) return;
    constexpr int PER = WIDTH / 32;
    float v[PER];
    float s = 0.0f;
    #pragma unroll
    for (int i = 0; i < PER; ++i) { v[i] = in[row * WIDTH + i * 32 + lane]; s += v[i]; }
    #pragma unroll
    for (int o = 16; o; o >>= 1) s += __shfl_xor(s, o, 32);
    float mean = s / (float)WIDTH;
    float q = 0.0f;
    #pragma unroll
    for (int i = 0; i < PER; ++i) { float d = v[i] - mean; q += d * d; }
    #pragma unroll
    for (int o = 16; o; o >>= 1) q += __shfl_xor(q, o, 32);
    float inv = rsqrtf(q / (float)WIDTH + EPSV);
    #pragma unroll
    for (int i = 0; i < PER; ++i) {
        int c = i * 32 + lane;
        float t = (v[i] - mean) * inv * g[c] + b[c];
        if (DOGELU) t = gelu_exact(t);
        out[row * WIDTH + c] = t;
    }
}

// ---------------------------------------------------------------------------
// Edge feature kernels
// stats: [0]=sum(dist) [1]=sum(1/dist) [2]=sum(dirx) [3]=sum(diry)
//        [4]=md  [5..8]=ea column means
// ---------------------------------------------------------------------------
__global__ void k_edge_stage1(const int* __restrict__ ei, const float* __restrict__ pos,
                              float* __restrict__ ea, int* __restrict__ src,
                              int* __restrict__ dst, float* __restrict__ stats)
{
    int e = blockIdx.x * blockDim.x + threadIdx.x;
    float d = 0.0f, invd = 0.0f, dx = 0.0f, dy = 0.0f;
    if (e < N_EDGES) {
        int r = ei[e], c = ei[N_EDGES + e];
        float ddx = pos[c * 2 + 0] - pos[r * 2 + 0];
        float ddy = pos[c * 2 + 1] - pos[r * 2 + 1];
        float dd = sqrtf(ddx * ddx + ddy * ddy);
        dd = fmaxf(dd, 1e-6f);
        dx = ddx / dd; dy = ddy / dd;
        d = dd; invd = 1.0f / dd;
        ea[e * 4 + 0] = dx;
        ea[e * 4 + 1] = dy;
        ea[e * 4 + 2] = dd;     // raw dist (finalized in stage2)
        src[e] = r; dst[e] = c;
    }
    #pragma unroll
    for (int o = 16; o; o >>= 1) {
        d    += __shfl_xor(d, o, 32);
        invd += __shfl_xor(invd, o, 32);
        dx   += __shfl_xor(dx, o, 32);
        dy   += __shfl_xor(dy, o, 32);
    }
    if ((threadIdx.x & 31) == 0) {
        atomicAdd(&stats[0], d);
        atomicAdd(&stats[1], invd);
        atomicAdd(&stats[2], dx);
        atomicAdd(&stats[3], dy);
    }
}

__global__ void k_finalize_stats(float* __restrict__ stats)
{
    float md = fmaxf(stats[0] / (float)N_EDGES, 1e-6f);
    stats[4] = md;
    stats[5] = stats[2] / (float)N_EDGES;
    stats[6] = stats[3] / (float)N_EDGES;
    stats[7] = (stats[0] / md) / (float)N_EDGES;
    stats[8] = (md * stats[1]) / (float)N_EDGES;
}

__global__ void k_edge_stage2(float* __restrict__ ea, const float* __restrict__ stats)
{
    int e = blockIdx.x * blockDim.x + threadIdx.x;
    if (e >= N_EDGES) return;
    float md = stats[4];
    float d = ea[e * 4 + 2];
    ea[e * 4 + 2] = d / md;
    ea[e * 4 + 3] = md / d;
}

__global__ void k_selfloops(float* __restrict__ ea, int* __restrict__ src,
                            int* __restrict__ dst, const float* __restrict__ stats)
{
    int i = blockIdx.x * blockDim.x + threadIdx.x;
    if (i >= N_NODES) return;
    src[N_EDGES + i] = i;
    dst[N_EDGES + i] = i;
    #pragma unroll
    for (int k = 0; k < 4; ++k) ea[(N_EDGES + i) * 4 + k] = stats[5 + k];
}

// ---------------------------------------------------------------------------
// GATv2 attention: one wave per (edge, head). ee = ea @ We folded in on-the-fly.
// ---------------------------------------------------------------------------
__launch_bounds__(256)
__global__ void k_att_score(const float* __restrict__ xl, const float* __restrict__ xr,
                            const float* __restrict__ ea, const float* __restrict__ We,
                            const float* __restrict__ att, const int* __restrict__ src,
                            const int* __restrict__ dst, float* __restrict__ aw,
                            unsigned* __restrict__ amaxkey)
{
    int gw = blockIdx.x * 8 + (threadIdx.x >> 5);
    if (gw >= ETOT * NHEAD) return;
    int e = gw >> 2, hd = gw & 3;
    int lane = threadIdx.x & 31;
    int s = src[e], dt = dst[e];
    float ea0 = ea[e * 4 + 0], ea1 = ea[e * 4 + 1];
    float ea2 = ea[e * 4 + 2], ea3 = ea[e * 4 + 3];
    float acc = 0.0f;
    #pragma unroll
    for (int j = 0; j < 4; ++j) {
        int c = hd * HID + j * 32 + lane;
        float eev = ea0 * We[c] + ea1 * We[512 + c] + ea2 * We[1024 + c] + ea3 * We[1536 + c];
        float m = xl[s * 512 + c] + xr[dt * 512 + c] + eev;
        m = (m > 0.0f) ? m : NEGS * m;
        acc += m * att[c];
    }
    #pragma unroll
    for (int o = 16; o; o >>= 1) acc += __shfl_xor(acc, o, 32);
    if (lane == 0) {
        aw[gw] = acc;
        atomicMax(&amaxkey[dt * NHEAD + hd], f2ord(acc));
    }
}

__global__ void k_att_norm(const int* __restrict__ dst, const unsigned* __restrict__ amaxkey,
                           float* __restrict__ aw, float* __restrict__ asum)
{
    int gw = blockIdx.x * blockDim.x + threadIdx.x;
    if (gw >= ETOT * NHEAD) return;
    int e = gw >> 2, hd = gw & 3;
    int dt = dst[e];
    float w = expf(aw[gw] - ord2f(amaxkey[dt * NHEAD + hd]));
    aw[gw] = w;
    atomicAdd(&asum[dt * NHEAD + hd], w);
}

__launch_bounds__(256)
__global__ void k_aggregate(const float* __restrict__ xl, const int* __restrict__ src,
                            const int* __restrict__ dst, const float* __restrict__ aw,
                            const float* __restrict__ asum, float* __restrict__ agg)
{
    int gw = blockIdx.x * 8 + (threadIdx.x >> 5);
    if (gw >= ETOT * NHEAD) return;
    int e = gw >> 2, hd = gw & 3;
    int lane = threadIdx.x & 31;
    int s = src[e], dt = dst[e];
    float alpha = aw[gw] / asum[dt * NHEAD + hd];
    #pragma unroll
    for (int j = 0; j < 4; ++j) {
        int c = hd * HID + j * 32 + lane;
        atomicAdd(&agg[dt * 512 + c], xl[s * 512 + c] * alpha);
    }
}

// ---------------------------------------------------------------------------
// GraphNorm pipeline
// ---------------------------------------------------------------------------
#define GN_ROWS 128
__launch_bounds__(128)
__global__ void k_heads_mean(const float* __restrict__ agg, const float* __restrict__ cb,
                             float* __restrict__ obuf, float* __restrict__ colsum)
{
    int c = threadIdx.x;
    int r0 = blockIdx.x * GN_ROWS;
    int r1 = r0 + GN_ROWS; if (r1 > N_NODES) r1 = N_NODES;
    float ps = 0.0f;
    for (int r = r0; r < r1; ++r) {
        float o = 0.25f * (agg[r * 512 + c] + agg[r * 512 + 128 + c] +
                           agg[r * 512 + 256 + c] + agg[r * 512 + 384 + c]) + cb[c];
        obuf[r * HID + c] = o;
        ps += o;
    }
    atomicAdd(&colsum[c], ps);
}

__global__ void k_mu_fin(const float* __restrict__ colsum, float* __restrict__ mu)
{
    int c = threadIdx.x;
    mu[c] = colsum[c] / (float)N_NODES;
}

__launch_bounds__(128)
__global__ void k_center_sq(float* __restrict__ obuf, const float* __restrict__ mu,
                            const float* __restrict__ ga, float* __restrict__ sqsum)
{
    int c = threadIdx.x;
    int r0 = blockIdx.x * GN_ROWS;
    int r1 = r0 + GN_ROWS; if (r1 > N_NODES) r1 = N_NODES;
    float m = ga[c] * mu[c];
    float ps = 0.0f;
    for (int r = r0; r < r1; ++r) {
        float t = obuf[r * HID + c] - m;
        obuf[r * HID + c] = t;
        ps += t * t;
    }
    atomicAdd(&sqsum[c], ps);
}

__global__ void k_scale_fin(const float* __restrict__ sqsum, const float* __restrict__ gw,
                            float* __restrict__ scale)
{
    int c = threadIdx.x;
    scale[c] = gw[c] * rsqrtf(sqsum[c] / (float)N_NODES + EPSV);
}

__global__ void k_gn_apply(const float* __restrict__ obuf, const float* __restrict__ scale,
                           const float* __restrict__ gb, const float* __restrict__ hold,
                           float* __restrict__ hnew)
{
    int idx = blockIdx.x * blockDim.x + threadIdx.x;
    if (idx >= N_NODES * HID) return;
    int c = idx & (HID - 1);
    hnew[idx] = gelu_exact(obuf[idx] * scale[c] + gb[c]) + hold[idx];
}

__global__ void k_gelu_elem(float* __restrict__ x, int n)
{
    int idx = blockIdx.x * blockDim.x + threadIdx.x;
    if (idx < n) x[idx] = gelu_exact(x[idx]);
}

// ---------------------------------------------------------------------------
extern "C" void kernel_launch(void* const* d_in, const int* in_sizes, int n_in,
                              void* d_out, int out_size, void* d_ws, size_t ws_size,
                              hipStream_t stream) {
    const float* x        = (const float*)d_in[0];
    const int*   ei       = (const int*)  d_in[1];
    const float* pos      = (const float*)d_in[2];
    const float* in_W     = (const float*)d_in[3];
    const float* in_b     = (const float*)d_in[4];
    const float* in_ln_g  = (const float*)d_in[5];
    const float* in_ln_b  = (const float*)d_in[6];
    const float* conv_Wl  = (const float*)d_in[7];
    const float* conv_bl  = (const float*)d_in[8];
    const float* conv_Wr  = (const float*)d_in[9];
    const float* conv_br  = (const float*)d_in[10];
    const float* conv_We  = (const float*)d_in[11];
    const float* conv_att = (const float*)d_in[12];
    const float* conv_b   = (const float*)d_in[13];
    const float* gn_w     = (const float*)d_in[14];
    const float* gn_b     = (const float*)d_in[15];
    const float* gn_a     = (const float*)d_in[16];
    const float* emb_W1   = (const float*)d_in[17];
    const float* emb_b1   = (const float*)d_in[18];
    const float* emb_ln1g = (const float*)d_in[19];
    const float* emb_ln1b = (const float*)d_in[20];
    const float* emb_W2   = (const float*)d_in[21];
    const float* emb_b2   = (const float*)d_in[22];
    const float* emb_ln2g = (const float*)d_in[23];
    const float* emb_ln2b = (const float*)d_in[24];
    const float* cls_W1   = (const float*)d_in[25];
    const float* cls_b1   = (const float*)d_in[26];
    const float* cls_W2   = (const float*)d_in[27];
    const float* cls_b2   = (const float*)d_in[28];
    float* out = (float*)d_out;

    // ---- carve workspace ----
    char* ws = (char*)d_ws;
    size_t off = 0;
    auto carve = [&](size_t bytes) -> char* {
        char* p = ws + off;
        off = (off + bytes + 255) & ~(size_t)255;
        return p;
    };
    float*    ea      = (float*)   carve((size_t)ETOT * 4 * 4);
    int*      srcA    = (int*)     carve((size_t)ETOT * 4);
    int*      dstA    = (int*)     carve((size_t)ETOT * 4);
    float*    stats   = (float*)   carve(16 * 4);
    float*    hA      = (float*)   carve((size_t)N_NODES * HID * 4);
    float*    hB      = (float*)   carve((size_t)N_NODES * HID * 4);
    float*    buf128  = (float*)   carve((size_t)N_NODES * HID * 4);   // pre-LN / GraphNorm scratch
    float*    xl      = (float*)   carve((size_t)N_NODES * 512 * 4);
    float*    xr      = (float*)   carve((size_t)N_NODES * 512 * 4);
    float*    aw      = (float*)   carve((size_t)ETOT * NHEAD * 4);
    unsigned* amaxkey = (unsigned*)carve((size_t)N_NODES * NHEAD * 4);
    float*    asum    = (float*)   carve((size_t)N_NODES * NHEAD * 4);
    float*    agg     = (float*)   carve((size_t)N_NODES * 512 * 4);
    float*    colsum  = (float*)   carve(HID * 4);
    float*    sqsum   = (float*)   carve(HID * 4);
    float*    mu      = (float*)   carve(HID * 4);
    float*    scale   = (float*)   carve(HID * 4);
    float*    buf64a  = (float*)   carve((size_t)N_NODES * EMBD * 4);  // pre-ln2 / classifier hidden
    float*    buf64b  = (float*)   carve((size_t)N_NODES * EMBD * 4);  // emb (ln2 output)

    auto gemm = [&](const float* A, const float* B, const float* bias, float* C,
                    int M, int K, int Nc) {
        dim3 g((M + 63) / 64, (Nc + 63) / 64);
        k_gemm_wmma<<<g, 128, 0, stream>>>(A, B, bias, C, M, K, Nc);
    };

    // ---- edge features ----
    hipMemsetAsync(stats, 0, 16 * 4, stream);
    k_edge_stage1<<<(N_EDGES + 255) / 256, 256, 0, stream>>>(ei, pos, ea, srcA, dstA, stats);
    k_finalize_stats<<<1, 1, 0, stream>>>(stats);
    k_edge_stage2<<<(N_EDGES + 255) / 256, 256, 0, stream>>>(ea, stats);
    k_selfloops<<<(N_NODES + 255) / 256, 256, 0, stream>>>(ea, srcA, dstA, stats);

    // ---- input projection: h = gelu(ln(x @ in_W + b)) ----
    gemm(x, in_W, in_b, buf128, N_NODES, DIN, HID);
    k_ln<HID, true><<<(N_NODES + 3) / 4, 128, 0, stream>>>(buf128, in_ln_g, in_ln_b, hA, N_NODES);

    float* cur = hA;
    float* alt = hB;
    const int nwaves = ETOT * NHEAD;

    for (int l = 0; l < NLAY; ++l) {
        const float* Wl = conv_Wl + (size_t)l * HID * 512;
        const float* bl = conv_bl + (size_t)l * 512;
        const float* Wr = conv_Wr + (size_t)l * HID * 512;
        const float* br = conv_br + (size_t)l * 512;
        const float* We = conv_We + (size_t)l * 4 * 512;
        const float* at = conv_att + (size_t)l * 512;
        const float* cb = conv_b  + (size_t)l * HID;

        gemm(cur, Wl, bl, xl, N_NODES, HID, 512);
        gemm(cur, Wr, br, xr, N_NODES, HID, 512);

        hipMemsetAsync(amaxkey, 0, (size_t)N_NODES * NHEAD * 4, stream);
        hipMemsetAsync(asum, 0, (size_t)N_NODES * NHEAD * 4, stream);
        hipMemsetAsync(agg, 0, (size_t)N_NODES * 512 * 4, stream);
        hipMemsetAsync(colsum, 0, HID * 4, stream);
        hipMemsetAsync(sqsum, 0, HID * 4, stream);

        k_att_score<<<(nwaves + 7) / 8, 256, 0, stream>>>(xl, xr, ea, We, at, srcA, dstA, aw, amaxkey);
        k_att_norm<<<(nwaves + 255) / 256, 256, 0, stream>>>(dstA, amaxkey, aw, asum);
        k_aggregate<<<(nwaves + 7) / 8, 256, 0, stream>>>(xl, srcA, dstA, aw, asum, agg);

        k_heads_mean<<<(N_NODES + GN_ROWS - 1) / GN_ROWS, 128, 0, stream>>>(agg, cb, buf128, colsum);
        k_mu_fin<<<1, 128, 0, stream>>>(colsum, mu);
        k_center_sq<<<(N_NODES + GN_ROWS - 1) / GN_ROWS, 128, 0, stream>>>(buf128, mu, gn_a + (size_t)l * HID, sqsum);
        k_scale_fin<<<1, 128, 0, stream>>>(sqsum, gn_w + (size_t)l * HID, scale);
        k_gn_apply<<<(N_NODES * HID + 255) / 256, 256, 0, stream>>>(buf128, scale, gn_b + (size_t)l * HID, cur, alt);

        float* t = cur; cur = alt; alt = t;
    }

    // ---- embedding head ----
    gemm(cur, emb_W1, emb_b1, buf128, N_NODES, HID, HID);
    k_ln<HID, true><<<(N_NODES + 3) / 4, 128, 0, stream>>>(buf128, emb_ln1g, emb_ln1b, alt, N_NODES);
    gemm(alt, emb_W2, emb_b2, buf64a, N_NODES, HID, EMBD);
    k_ln<EMBD, false><<<(N_NODES + 3) / 4, 128, 0, stream>>>(buf64a, emb_ln2g, emb_ln2b, buf64b, N_NODES);

    // ---- classifier ----
    gemm(buf64b, cls_W1, cls_b1, buf64a, N_NODES, EMBD, EMBD);
    k_gelu_elem<<<(N_NODES * EMBD + 255) / 256, 256, 0, stream>>>(buf64a, N_NODES * EMBD);
    gemm(buf64a, cls_W2, cls_b2, out, N_NODES, EMBD, NCLS);
}
